// CenterPointHead_62096637165646
// MI455X (gfx1250) — matrix operand order repaired and checked by reading, not compiled, not run on hardware
//
#include <hip/hip_runtime.h>

// ---------------------------------------------------------------------------
// CenterPoint head for MI455X (gfx1250, wave32, WMMA bf16)
//   prep: weights -> bf16 staging layout + BN constants (in d_ws)
//   A   : conv3x3(256->64)+BN+ReLU -> shared activation (bf16 NHWC in d_ws)
//   B x2: conv3x3(64->64)+BN+ReLU -> conv1x1(64->NC)+bias  (all WMMA)
// ---------------------------------------------------------------------------

typedef __attribute__((ext_vector_type(16))) __bf16 v16bf;
typedef __attribute__((ext_vector_type(8)))  float  v8f;
typedef __attribute__((ext_vector_type(4)))  int    i32x4;
typedef __attribute__((ext_vector_type(2)))  int    i32x2;

union Frag { v16bf v; unsigned u[8]; };

#define HW   300
#define TN   128
#define EPSI 1e-5f

// ---- CDNA5 async global->LDS path (guarded; falls back to plain copies) ----
#if defined(__has_builtin)
#  if __has_builtin(__builtin_amdgcn_global_load_async_to_lds_b128) && \
      __has_builtin(__builtin_amdgcn_global_load_async_to_lds_b64)  && \
      __has_builtin(__builtin_amdgcn_s_wait_asynccnt)
#    define USE_ASYNC_LDS 1
#  endif
#endif

#define GLB_AS __attribute__((address_space(1)))
#define LDS_AS __attribute__((address_space(3)))

__device__ __forceinline__ void ld_lds_b128(void* dst, const void* src) {
#ifdef USE_ASYNC_LDS
    __builtin_amdgcn_global_load_async_to_lds_b128(
        (GLB_AS i32x4*)(void*)src, (LDS_AS i32x4*)dst, 0, 0);
#else
    *(uint4*)dst = *(const uint4*)src;
#endif
}
__device__ __forceinline__ void ld_lds_b64(void* dst, const void* src) {
#ifdef USE_ASYNC_LDS
    __builtin_amdgcn_global_load_async_to_lds_b64(
        (GLB_AS i32x2*)(void*)src, (LDS_AS i32x2*)dst, 0, 0);
#else
    *(unsigned long long*)dst = *(const unsigned long long*)src;
#endif
}
__device__ __forceinline__ void stage_fence() {
#ifdef USE_ASYNC_LDS
    __builtin_amdgcn_s_wait_asynccnt(0);
#endif
}

__device__ __forceinline__ unsigned short f2bf(float f) {
    union { float f; unsigned u; } x; x.f = f;
    unsigned r = x.u + 0x7FFFu + ((x.u >> 16) & 1u);   // RNE
    return (unsigned short)(r >> 16);
}
// ISA 16-bit A 16x32 layout: lanes0-15 K-base 0, 16-31 base 8; VGPR4-7 at K+16
__device__ __forceinline__ int a_kmap(int v, int hi) {
    return (hi ? 8 : 0) + ((v < 4) ? 2 * v : 16 + 2 * (v - 4));
}
// B 32x16 layout matching A: lanes0-15 K-base 0, 16-31 base 8; VGPR4-7 at K+16
__device__ __forceinline__ int b_kmap(int v, int hi) {
    return (hi ? 8 : 0) + ((v < 4) ? 2 * v : 16 + 2 * (v - 4));
}

// ============== prep: weights->bf16 staging layout, BN constants ============
// wA:[8][9][64][32]  wH/wR:[2][9][64][32]  w2H/w2R:[16][64]  bn:[3][2][64]
__global__ __launch_bounds__(256) void prep_kernel(
    const float* __restrict__ w_sh,  const float* __restrict__ w_hm1,
    const float* __restrict__ w_rg1, const float* __restrict__ w_hm2,
    const float* __restrict__ w_rg2,
    const float* __restrict__ g0, const float* __restrict__ b0,
    const float* __restrict__ m0, const float* __restrict__ v0,
    const float* __restrict__ g1, const float* __restrict__ b1,
    const float* __restrict__ m1, const float* __restrict__ v1,
    const float* __restrict__ g2, const float* __restrict__ b2,
    const float* __restrict__ m2, const float* __restrict__ v2,
    unsigned short* __restrict__ wA, unsigned short* __restrict__ wH,
    unsigned short* __restrict__ wR, unsigned short* __restrict__ w2H,
    unsigned short* __restrict__ w2R, float* __restrict__ bn)
{
    int id = blockIdx.x * 256 + threadIdx.x;
    if (id < 147456) {                                  // [kc][t][o][c] <- [o][256][3][3]
        int c = id & 31, r = id >> 5, o = r & 63; r >>= 6;
        int t = r % 9, kc = r / 9;
        wA[id] = f2bf(w_sh[((o * 256) + kc * 32 + c) * 9 + t]);
        return;
    }
    id -= 147456;
    if (id < 36864) {
        int c = id & 31, r = id >> 5, o = r & 63; r >>= 6;
        int t = r % 9, kc = r / 9;
        wH[id] = f2bf(w_hm1[((o * 64) + kc * 32 + c) * 9 + t]);
        return;
    }
    id -= 36864;
    if (id < 36864) {
        int c = id & 31, r = id >> 5, o = r & 63; r >>= 6;
        int t = r % 9, kc = r / 9;
        wR[id] = f2bf(w_rg1[((o * 64) + kc * 32 + c) * 9 + t]);
        return;
    }
    id -= 36864;
    if (id < 1024) { int m = id >> 6, k = id & 63; w2H[id] = f2bf(m < 3 ? w_hm2[m * 64 + k] : 0.f); return; }
    id -= 1024;
    if (id < 1024) { int m = id >> 6, k = id & 63; w2R[id] = f2bf(m < 8 ? w_rg2[m * 64 + k] : 0.f); return; }
    id -= 1024;
    if (id < 192) {
        int h = id >> 6, o = id & 63;
        const float* g  = h == 0 ? g0 : (h == 1 ? g1 : g2);
        const float* bb = h == 0 ? b0 : (h == 1 ? b1 : b2);
        const float* mm = h == 0 ? m0 : (h == 1 ? m1 : m2);
        const float* vv = h == 0 ? v0 : (h == 1 ? v1 : v2);
        float sc = g[o] * rsqrtf(vv[o] + EPSI);
        bn[h * 128 + o]      = sc;
        bn[h * 128 + 64 + o] = bb[o] - mm[o] * sc;
    }
}

// ======================= Kernel A: conv_shared =============================
__global__ __launch_bounds__(256) void conv_shared_kernel(
    const float* __restrict__ in, const unsigned short* __restrict__ wA,
    const float* __restrict__ bn, unsigned short* __restrict__ out)
{
    __shared__ __align__(16) unsigned short sIn[3 * 130 * 34];  // [row][col][c], stride 34
    __shared__ __align__(16) unsigned short sW [9 * 64 * 32];   // [tap][o][c]
    __shared__ float sScale[64], sShift[64];

    const int tid = threadIdx.x;
    const int x0 = blockIdx.x * TN, y = blockIdx.y, b = blockIdx.z;
    if (tid < 64) { sScale[tid] = bn[tid]; sShift[tid] = bn[64 + tid]; }

    const int lane = tid & 31, wv = tid >> 5;
    const int wm = wv & 1, wn = wv >> 1;
    const int hi = lane >> 4, lm = lane & 15;

    v8f acc[2][2] = {};

    for (int kc = 0; kc < 8; ++kc) {
        __syncthreads();
        // ---- input slab: coalesced fp32 loads (col fastest), convert to bf16
        for (int i = tid; i < 3 * 130 * 32; i += 256) {
            int col = i % 130, r = i / 130;
            int row = r % 3,   c = r / 3;
            int gy = y + row - 1, gx = x0 + col - 1;
            float val = 0.f;
            if ((unsigned)gy < HW && (unsigned)gx < HW)
                val = in[((b * 256 + kc * 32 + c) * HW + gy) * HW + gx];
            sIn[(row * 130 + col) * 34 + c] = f2bf(val);
        }
        // ---- weight slab: contiguous bf16, async b128 to LDS
        const unsigned short* wp = wA + kc * (9 * 64 * 32);
        for (int i = tid; i < (9 * 64 * 32) / 8; i += 256)
            ld_lds_b128(&sW[i * 8], &wp[i * 8]);
        stage_fence();
        __syncthreads();

        // ---- prefetch next chunk's input rows while computing
        if (kc + 1 < 8) {
            for (int i = tid; i < 96 * 5; i += 256) {
                int s = i / 5, seg = i % 5;
                int c = s / 3, row = s % 3;
                int gy = y + row - 1;
                int gx = x0 - 1 + seg * 32; if (gx < 0) gx = 0;
                if ((unsigned)gy < HW && gx < HW)
                    __builtin_prefetch(&in[((b * 256 + (kc + 1) * 32 + c) * HW + gy) * HW + gx], 0, 3);
            }
        }

        #pragma unroll
        for (int t = 0; t < 9; ++t) {
            const int dy = t / 3, dx = t % 3;
            Frag a[2], bf[2];
            #pragma unroll
            for (int mt = 0; mt < 2; ++mt) {
                const int mrow = wm * 32 + mt * 16 + lm;
                #pragma unroll
                for (int v = 0; v < 8; ++v)
                    a[mt].u[v] = *(const unsigned*)&sW[(t * 64 + mrow) * 32 + a_kmap(v, hi)];
            }
            #pragma unroll
            for (int nt = 0; nt < 2; ++nt) {
                const int col = wn * 32 + nt * 16 + lm + dx;
                #pragma unroll
                for (int v = 0; v < 8; ++v)
                    bf[nt].u[v] = *(const unsigned*)&sIn[(dy * 130 + col) * 34 + b_kmap(v, hi)];
            }
            #pragma unroll
            for (int mt = 0; mt < 2; ++mt)
                #pragma unroll
                for (int nt = 0; nt < 2; ++nt)
                    acc[mt][nt] = __builtin_amdgcn_wmma_f32_16x16x32_bf16(
                        false, a[mt].v, false, bf[nt].v, (short)0, acc[mt][nt], false, false);
        }
    }

    #pragma unroll
    for (int mt = 0; mt < 2; ++mt)
        #pragma unroll
        for (int nt = 0; nt < 2; ++nt)
            #pragma unroll
            for (int v = 0; v < 8; ++v) {
                int o = wm * 32 + mt * 16 + v + hi * 8;
                int x = x0 + wn * 32 + nt * 16 + lm;
                float val = acc[mt][nt][v] * sScale[o] + sShift[o];
                val = val > 0.f ? val : 0.f;
                if (x < HW)
                    out[((b * HW + y) * HW + x) * 64 + o] = f2bf(val);
            }
}

// ======================= Kernel B: one detection head ======================
__global__ __launch_bounds__(256) void head_kernel(
    const unsigned short* __restrict__ in, const unsigned short* __restrict__ wP,
    const unsigned short* __restrict__ w2P, const float* __restrict__ bn,
    const float* __restrict__ b2, float* __restrict__ out, int NC)
{
    __shared__ __align__(16) unsigned short sIn[3 * 130 * 36];  // stride 36 (8B-aligned rows)
    __shared__ __align__(16) unsigned short sW [9 * 64 * 32];
    __shared__ float sScale[64], sShift[64];

    const int tid = threadIdx.x;
    const int x0 = blockIdx.x * TN, y = blockIdx.y, b = blockIdx.z;
    if (tid < 64) { sScale[tid] = bn[tid]; sShift[tid] = bn[64 + tid]; }

    const int lane = tid & 31, wv = tid >> 5;
    const int wm = wv & 1, wn = wv >> 1;
    const int hi = lane >> 4, lm = lane & 15;

    v8f acc[2][2] = {};

    for (int kc = 0; kc < 2; ++kc) {
        __syncthreads();
        // ---- activation slab (bf16): async b64 quarters, zero-fill OOB
        for (int i = tid; i < 390 * 8; i += 256) {
            int q = i & 7, p = i >> 3;
            int col = p % 130, row = p / 130;
            int gy = y + row - 1, gx = x0 + col - 1;
            unsigned short* dst = &sIn[p * 36 + q * 4];
            if ((unsigned)gy < HW && (unsigned)gx < HW)
                ld_lds_b64(dst, &in[((b * HW + gy) * HW + gx) * 64 + kc * 32 + q * 4]);
            else
                *(unsigned long long*)dst = 0ull;
        }
        // ---- weight slab
        const unsigned short* wp = wP + kc * (9 * 64 * 32);
        for (int i = tid; i < (9 * 64 * 32) / 8; i += 256)
            ld_lds_b128(&sW[i * 8], &wp[i * 8]);
        stage_fence();
        __syncthreads();

        #pragma unroll
        for (int t = 0; t < 9; ++t) {
            const int dy = t / 3, dx = t % 3;
            Frag a[2], bf[2];
            #pragma unroll
            for (int mt = 0; mt < 2; ++mt) {
                const int mrow = wm * 32 + mt * 16 + lm;
                #pragma unroll
                for (int v = 0; v < 8; ++v)
                    a[mt].u[v] = *(const unsigned*)&sW[(t * 64 + mrow) * 32 + a_kmap(v, hi)];
            }
            #pragma unroll
            for (int nt = 0; nt < 2; ++nt) {
                const int col = wn * 32 + nt * 16 + lm + dx;
                #pragma unroll
                for (int v = 0; v < 8; ++v)
                    bf[nt].u[v] = *(const unsigned*)&sIn[(dy * 130 + col) * 36 + b_kmap(v, hi)];
            }
            #pragma unroll
            for (int mt = 0; mt < 2; ++mt)
                #pragma unroll
                for (int nt = 0; nt < 2; ++nt)
                    acc[mt][nt] = __builtin_amdgcn_wmma_f32_16x16x32_bf16(
                        false, a[mt].v, false, bf[nt].v, (short)0, acc[mt][nt], false, false);
        }
    }

    // ---- BN+ReLU -> LDS [pos][ch] (stride 68), stage padded 1x1 weights
    __syncthreads();
    unsigned short* sHM = sIn;     // 128*68 = 8704 <= 14040
    unsigned short* sW2 = sW;      // 1024   <= 18432
    #pragma unroll
    for (int mt = 0; mt < 2; ++mt)
        #pragma unroll
        for (int nt = 0; nt < 2; ++nt)
            #pragma unroll
            for (int v = 0; v < 8; ++v) {
                int o = wm * 32 + mt * 16 + v + hi * 8;
                int n = wn * 32 + nt * 16 + lm;
                float val = acc[mt][nt][v] * sScale[o] + sShift[o];
                val = val > 0.f ? val : 0.f;
                sHM[n * 68 + o] = f2bf(val);
            }
    for (int i = tid; i < 1024 / 8; i += 256)
        ld_lds_b128(&sW2[i * 8], &w2P[i * 8]);
    stage_fence();
    __syncthreads();

    // ---- 1x1 head as WMMA: M=16(pad NC), N=16 pos/wave, K=64
    v8f acc1 = {};
    #pragma unroll
    for (int ks = 0; ks < 2; ++ks) {
        Frag a, bf;
        #pragma unroll
        for (int v = 0; v < 8; ++v) {
            a.u[v]  = *(const unsigned*)&sW2[lm * 64 + ks * 32 + a_kmap(v, hi)];
            bf.u[v] = *(const unsigned*)&sHM[(wv * 16 + lm) * 68 + ks * 32 + b_kmap(v, hi)];
        }
        acc1 = __builtin_amdgcn_wmma_f32_16x16x32_bf16(
            false, a.v, false, bf.v, (short)0, acc1, false, false);
    }
    #pragma unroll
    for (int v = 0; v < 8; ++v) {
        int nc = v + hi * 8;
        int x  = x0 + wv * 16 + lm;
        if (nc < NC && x < HW)
            out[((b * NC + nc) * HW + y) * HW + x] = acc1[v] + b2[nc];
    }
}

// ============================== launch =====================================
#define OFS_WA  46080000u
#define OFS_WH  46374912u
#define OFS_WR  46448640u
#define OFS_W2H 46522368u
#define OFS_W2R 46524416u
#define OFS_BN  46526464u

extern "C" void kernel_launch(void* const* d_in, const int* in_sizes, int n_in,
                              void* d_out, int out_size, void* d_ws, size_t ws_size,
                              hipStream_t stream) {
    const float* bev   = (const float*)d_in[0];
    const float* w_sh  = (const float*)d_in[1];
    const float* g_sh  = (const float*)d_in[2];
    const float* b_sh  = (const float*)d_in[3];
    const float* m_sh  = (const float*)d_in[4];
    const float* v_sh  = (const float*)d_in[5];
    const float* w_hm1 = (const float*)d_in[6];
    const float* g_hm1 = (const float*)d_in[7];
    const float* b_hm1 = (const float*)d_in[8];
    const float* m_hm1 = (const float*)d_in[9];
    const float* v_hm1 = (const float*)d_in[10];
    const float* w_hm2 = (const float*)d_in[11];
    const float* b_hm2 = (const float*)d_in[12];
    const float* w_rg1 = (const float*)d_in[13];
    const float* g_rg1 = (const float*)d_in[14];
    const float* b_rg1 = (const float*)d_in[15];
    const float* m_rg1 = (const float*)d_in[16];
    const float* v_rg1 = (const float*)d_in[17];
    const float* w_rg2 = (const float*)d_in[18];
    const float* b_rg2 = (const float*)d_in[19];

    unsigned char* ws = (unsigned char*)d_ws;
    unsigned short* shared = (unsigned short*)ws;          // [4,300,300,64] bf16
    unsigned short* wA  = (unsigned short*)(ws + OFS_WA);
    unsigned short* wH  = (unsigned short*)(ws + OFS_WH);
    unsigned short* wR  = (unsigned short*)(ws + OFS_WR);
    unsigned short* w2H = (unsigned short*)(ws + OFS_W2H);
    unsigned short* w2R = (unsigned short*)(ws + OFS_W2R);
    float*          bn  = (float*)         (ws + OFS_BN);  // [3][2][64]

    float* heatmap = (float*)d_out;
    float* box_reg = heatmap + 4 * 3 * HW * HW;

    prep_kernel<<<dim3(873), dim3(256), 0, stream>>>(
        w_sh, w_hm1, w_rg1, w_hm2, w_rg2,
        g_sh, b_sh, m_sh, v_sh,
        g_hm1, b_hm1, m_hm1, v_hm1,
        g_rg1, b_rg1, m_rg1, v_rg1,
        wA, wH, wR, w2H, w2R, bn);

    dim3 grid((HW + TN - 1) / TN, HW, 4);
    dim3 block(256);
    conv_shared_kernel<<<grid, block, 0, stream>>>(bev, wA, bn, shared);
    head_kernel<<<grid, block, 0, stream>>>(shared, wH, w2H, bn + 128, b_hm2, heatmap, 3);
    head_kernel<<<grid, block, 0, stream>>>(shared, wR, w2R, bn + 256, b_rg2, box_reg, 8);
}
